// Decoder_69544110457296
// MI455X (gfx1250) — matrix operand order repaired and checked by reading, not compiled, not run on hardware
//
#include <hip/hip_runtime.h>

typedef _Float16 h16;
typedef __attribute__((ext_vector_type(16))) _Float16 v16h;
typedef __attribute__((ext_vector_type(8)))  _Float16 v8h;
typedef __attribute__((ext_vector_type(8)))  float    v8f;

#define NB    64      // batch
#define TT    512     // encoder timesteps
#define KEYD  128
#define VALD  128
#define HID   256
#define LSTEP 256     // decode length
#define VOC   4096

// ---------------- workspace layout (bytes, all 32B aligned) ----------------
#define OFF_WIH1 0u
#define SZ_WIH1  (12u*64u*32u*16u*2u)    // W_ih1^T frags: KT=12, NT=64
#define OFF_WHH1 (OFF_WIH1 + SZ_WIH1)
#define SZ_WHH1  (8u*64u*32u*16u*2u)     // W_hh1^T frags: KT=8, NT=64
#define OFF_WIH2 (OFF_WHH1 + SZ_WHH1)
#define SZ_WIH2  (8u*32u*32u*16u*2u)     // W_ih2^T frags: KT=8, NT=32
#define OFF_WHH2 (OFF_WIH2 + SZ_WIH2)
#define SZ_WHH2  (4u*32u*32u*16u*2u)     // W_hh2^T frags: KT=4, NT=32
#define OFF_EPK  (OFF_WHH2 + SZ_WHH2)
#define SZ_EPK   (8u*256u*32u*16u*2u)    // embedding^T frags: KT=8, NT=256
#define OFF_B1   (OFF_EPK + SZ_EPK)
#define SZ_B1    (1024u*4u)
#define OFF_B2   (OFF_B1 + SZ_B1)
#define SZ_B2    (512u*4u)
#define OFF_S    (OFF_B2 + SZ_B2)
#define SZ_S     (16384u*256u*2u)        // concat(h2,ctx) per (b,l), f16

__device__ __forceinline__ float sigmoidf_(float x){ return 1.0f/(1.0f + __expf(-x)); }

__device__ __forceinline__ v8f wmma16(v16h a, v16h b, v8f c){
  return __builtin_amdgcn_wmma_f32_16x16x32_f16(false, a, false, b, (short)0, c, false, false);
}

// A fragment (16-bit A 16x32 layout, ISA 7.12.2) where every lane reads the SAME
// activation row (matvec: all 16 M rows of D become valid duplicates).
__device__ __forceinline__ v16h row_afrag(const h16* __restrict__ row, int kt, int lane){
  const h16* p = row + kt*32 + ((lane >> 4) << 3);   // kb = (lane>=16)?8:0
  v8h lo = *(const v8h*)p;          // K = kt*32+kb .. +7
  v8h hi = *(const v8h*)(p + 16);   // K = kt*32+kb+16 .. +23
  return __builtin_shufflevector(lo, hi, 0,1,2,3,4,5,6,7,8,9,10,11,12,13,14,15);
}

__device__ __forceinline__ v16h ldB(const h16* __restrict__ base, int kt, int NT,
                                    int ct, int lane){
  return *(const v16h*)(base + (((size_t)kt*NT + ct)*32 + lane)*16);
}

// ---------------- phase 0: pack W[n][k] (f32) -> B-fragment layout (f16) ----
// dst[kt][nt][lane][j] = W[n][k], k = kt*32 + (lane>=16)*8 + (j&7) + (j>=8)*16, n = nt*16 + (lane&15)
__global__ __launch_bounds__(256) void pack_b_frag(const float* __restrict__ W,
                                                   h16* __restrict__ dst,
                                                   int K, int N){
  int idx = blockIdx.x*256 + threadIdx.x;
  int total = (K/32)*(N/16)*512;
  if (idx >= total) return;
  int j    = idx & 15;
  int lane = (idx >> 4) & 31;
  int tile = idx >> 9;
  int NT = N >> 4;
  int nt = tile % NT, kt = tile / NT;
  int k = kt*32 + ((lane >> 4) << 3) + (j & 7) + ((j >> 3) << 4);
  int n = nt*16 + (lane & 15);
  dst[idx] = (h16)W[(size_t)n*K + k];
}

__global__ __launch_bounds__(256) void bias_combine(const float* __restrict__ bi1,
                                                    const float* __restrict__ bh1,
                                                    const float* __restrict__ bi2,
                                                    const float* __restrict__ bh2,
                                                    float* __restrict__ b1,
                                                    float* __restrict__ b2){
  int i = blockIdx.x*256 + threadIdx.x;
  if (i < 1024) b1[i] = bi1[i] + bh1[i];
  if (i < 512)  b2[i] = bi2[i] + bh2[i];
}

// ---------------- phase 1: per-batch recurrent decoder ----------------------
// grid = 64 (one WG per batch element), block = 256 (8 waves). No inter-WG sync.
// key/values for this b live in LDS for the whole kernel; weights stream from L2.
__global__ __launch_bounds__(256) void decoder_recurrent(
    const float* __restrict__ key,     // [64][512][128] f32
    const float* __restrict__ values,  // [64][512][128] f32
    const int*   __restrict__ text,    // [64][256]
    const float* __restrict__ emb,     // [4096][256] f32
    const h16*   __restrict__ wih1f,   // [12][64][32][16]
    const h16*   __restrict__ whh1f,   // [8][64][32][16]
    const h16*   __restrict__ wih2f,   // [8][32][32][16]
    const h16*   __restrict__ whh2f,   // [4][32][32][16]
    const float* __restrict__ b1,      // [1024] (i,f,g,o)
    const float* __restrict__ b2,      // [512]
    h16*         __restrict__ S)       // [64*256][256]  concat(h2,ctx)
{
  __shared__ __align__(32) h16 sKeyT[KEYD*TT];   // keyT[k][t]   128KB
  __shared__ __align__(32) h16 sVals[TT*VALD];   // vals[t][v]   128KB
  __shared__ __align__(32) h16 sInp[HID+VALD];   // [emb | ctx]
  __shared__ __align__(32) h16 sH1[HID];
  __shared__ __align__(32) h16 sH2[KEYD];
  __shared__ float sAttn[TT];
  __shared__ float sRedA[8], sRedB[8];
  __shared__ float sPctx[256];

  const int b    = blockIdx.x;
  const int tid  = threadIdx.x;
  const int w    = tid >> 5;
  const int lane = tid & 31;

  // --- preload this batch's key (transposed) and values into LDS as f16 ---
  const float* kb_ = key    + (size_t)b*TT*KEYD;
  const float* vb_ = values + (size_t)b*TT*VALD;
  for (int i = tid; i < TT*KEYD; i += 256){
    int t = i >> 7, k = i & 127;
    sKeyT[k*TT + t] = (h16)kb_[i];
    sVals[i]        = (h16)vb_[i];
  }
  // --- zero initial state ---
  for (int i = tid; i < HID+VALD; i += 256) sInp[i] = (h16)0.0f;
  for (int i = tid; i < HID;      i += 256) sH1[i]  = (h16)0.0f;
  for (int i = tid; i < KEYD;     i += 256) sH2[i]  = (h16)0.0f;

  float c1[2] = {0.0f, 0.0f};  // cell state, lanes 0..15 only (h = (2w+ut)*16+lane)
  float c2    = 0.0f;          // lanes 0..15 only (h = w*16+lane)
  const v8f zero8 = {0.f,0.f,0.f,0.f,0.f,0.f,0.f,0.f};

  __syncthreads();

  for (int l = 0; l < LSTEP; ++l){
    // Opaque zero regenerated every step: weight-fragment addresses become
    // loop-variant so LICM cannot hoist (and spill) the B-fragment loads out
    // of the 256-step loop. They must stream from (L2-resident) memory.
    int opq;
    asm volatile("s_mov_b32 %0, 0" : "=s"(opq));
    const h16* wih1p = wih1f + opq;
    const h16* whh1p = whh1f + opq;
    const h16* wih2p = wih2f + opq;
    const h16* whh2p = whh2f + opq;

    // ---- 1) embedding lookup into sInp[0..255] ----
    int tok = text[b*LSTEP + l];
    sInp[tid] = (h16)emb[(size_t)tok*HID + tid];
    __syncthreads();

    // ---- 2) LSTM1 gates: g = inp@Wih1^T + h1@Whh1^T  (wave owns hidden tiles 2w,2w+1) ----
    v8f acc1[2][4];
    #pragma unroll
    for (int ut = 0; ut < 2; ++ut)
      #pragma unroll
      for (int g = 0; g < 4; ++g) acc1[ut][g] = zero8;

    #pragma unroll 2
    for (int ut = 0; ut < 2; ++ut){
      int nh = 2*w + ut;   // hidden tile 0..15
      for (int kt = 0; kt < 12; ++kt){
        v16h a  = row_afrag(sInp, kt, lane);
        v16h b0 = ldB(wih1p, kt, 64, 0*16 + nh, lane);
        v16h b1_ = ldB(wih1p, kt, 64, 1*16 + nh, lane);
        v16h b2_ = ldB(wih1p, kt, 64, 2*16 + nh, lane);
        v16h b3 = ldB(wih1p, kt, 64, 3*16 + nh, lane);
        acc1[ut][0] = wmma16(a, b0, acc1[ut][0]);
        acc1[ut][1] = wmma16(a, b1_, acc1[ut][1]);
        acc1[ut][2] = wmma16(a, b2_, acc1[ut][2]);
        acc1[ut][3] = wmma16(a, b3, acc1[ut][3]);
      }
      for (int kt = 0; kt < 8; ++kt){
        v16h a  = row_afrag(sH1, kt, lane);
        v16h b0 = ldB(whh1p, kt, 64, 0*16 + nh, lane);
        v16h b1_ = ldB(whh1p, kt, 64, 1*16 + nh, lane);
        v16h b2_ = ldB(whh1p, kt, 64, 2*16 + nh, lane);
        v16h b3 = ldB(whh1p, kt, 64, 3*16 + nh, lane);
        acc1[ut][0] = wmma16(a, b0, acc1[ut][0]);
        acc1[ut][1] = wmma16(a, b1_, acc1[ut][1]);
        acc1[ut][2] = wmma16(a, b2_, acc1[ut][2]);
        acc1[ut][3] = wmma16(a, b3, acc1[ut][3]);
      }
    }
    __syncthreads();   // all reads of old sH1 done

    // ---- 3) LSTM1 elementwise update, write new h1 ----
    if (lane < 16){
      #pragma unroll
      for (int ut = 0; ut < 2; ++ut){
        int h = (2*w + ut)*16 + lane;
        float gi = acc1[ut][0][0] + b1[0*HID + h];
        float gf = acc1[ut][1][0] + b1[1*HID + h];
        float gg = acc1[ut][2][0] + b1[2*HID + h];
        float go = acc1[ut][3][0] + b1[3*HID + h];
        float c  = sigmoidf_(gf)*c1[ut] + sigmoidf_(gi)*tanhf(gg);
        c1[ut] = c;
        sH1[h] = (h16)(sigmoidf_(go)*tanhf(c));
      }
    }
    __syncthreads();

    // ---- 4) LSTM2 gates: h1@Wih2^T + h2@Whh2^T (wave owns hidden tile w) ----
    v8f acc2[4];
    #pragma unroll
    for (int g = 0; g < 4; ++g) acc2[g] = zero8;
    for (int kt = 0; kt < 8; ++kt){
      v16h a  = row_afrag(sH1, kt, lane);
      v16h b0 = ldB(wih2p, kt, 32, 0*8 + w, lane);
      v16h b1_ = ldB(wih2p, kt, 32, 1*8 + w, lane);
      v16h b2_ = ldB(wih2p, kt, 32, 2*8 + w, lane);
      v16h b3 = ldB(wih2p, kt, 32, 3*8 + w, lane);
      acc2[0] = wmma16(a, b0, acc2[0]);
      acc2[1] = wmma16(a, b1_, acc2[1]);
      acc2[2] = wmma16(a, b2_, acc2[2]);
      acc2[3] = wmma16(a, b3, acc2[3]);
    }
    for (int kt = 0; kt < 4; ++kt){
      v16h a  = row_afrag(sH2, kt, lane);
      v16h b0 = ldB(whh2p, kt, 32, 0*8 + w, lane);
      v16h b1_ = ldB(whh2p, kt, 32, 1*8 + w, lane);
      v16h b2_ = ldB(whh2p, kt, 32, 2*8 + w, lane);
      v16h b3 = ldB(whh2p, kt, 32, 3*8 + w, lane);
      acc2[0] = wmma16(a, b0, acc2[0]);
      acc2[1] = wmma16(a, b1_, acc2[1]);
      acc2[2] = wmma16(a, b2_, acc2[2]);
      acc2[3] = wmma16(a, b3, acc2[3]);
    }
    __syncthreads();   // all reads of old sH2 done

    // ---- 5) LSTM2 update, write new h2 ----
    if (lane < 16){
      int h = w*16 + lane;
      float gi = acc2[0][0] + b2[0*KEYD + h];
      float gf = acc2[1][0] + b2[1*KEYD + h];
      float gg = acc2[2][0] + b2[2*KEYD + h];
      float go = acc2[3][0] + b2[3*KEYD + h];
      float c  = sigmoidf_(gf)*c2 + sigmoidf_(gi)*tanhf(gg);
      c2 = c;
      sH2[h] = (h16)(sigmoidf_(go)*tanhf(c));
    }
    __syncthreads();

    // ---- 6) attention: energy = key . h2, softmax over T ----
    float e0 = 0.0f, e1 = 0.0f;
    for (int k = 0; k < KEYD; ++k){
      float hv = (float)sH2[k];                 // LDS broadcast
      e0 += hv * (float)sKeyT[k*TT + tid];
      e1 += hv * (float)sKeyT[k*TT + tid + 256];
    }
    float mx = fmaxf(e0, e1);
    for (int off = 16; off; off >>= 1) mx = fmaxf(mx, __shfl_xor(mx, off, 32));
    if (lane == 0) sRedA[w] = mx;
    __syncthreads();
    mx = sRedA[0];
    #pragma unroll
    for (int i = 1; i < 8; ++i) mx = fmaxf(mx, sRedA[i]);
    float x0 = __expf(e0 - mx), x1 = __expf(e1 - mx);
    float sm = x0 + x1;
    for (int off = 16; off; off >>= 1) sm += __shfl_xor(sm, off, 32);
    if (lane == 0) sRedB[w] = sm;
    __syncthreads();
    sm = 0.0f;
    #pragma unroll
    for (int i = 0; i < 8; ++i) sm += sRedB[i];
    float inv = 1.0f / sm;
    sAttn[tid]       = x0 * inv;
    sAttn[tid + 256] = x1 * inv;
    __syncthreads();

    // ---- 7) ctx[v] = sum_t attn[t]*values[t][v]  (split t across 2 halves) ----
    {
      int v  = tid & 127;
      int th = tid >> 7;
      float acc = 0.0f;
      int t0 = th*256;
      for (int t = t0; t < t0 + 256; ++t)
        acc += sAttn[t] * (float)sVals[t*VALD + v];
      sPctx[tid] = acc;
    }
    __syncthreads();
    {
      h16* Srow = S + (size_t)(b*LSTEP + l)*256;
      if (tid < 128){
        float ctx = sPctx[tid] + sPctx[128 + tid];
        h16 ch = (h16)ctx;
        sInp[HID + tid] = ch;        // ctx feeds next step's LSTM1 input
        Srow[128 + tid] = ch;        // logits input: [h2 | ctx]
      } else {
        int j = tid - 128;
        Srow[j] = sH2[j];
      }
    }
    __syncthreads();
  }
}

// ---------------- phase 2: logits GEMM  out[16384,4096] = S @ E^T + b_out ----
// grid = (1024 Mtiles, 32), block = 256 (8 waves); wave -> one 16x16 tile, K=256.
__global__ __launch_bounds__(256) void decoder_logits(
    const h16*   __restrict__ S,     // [16384][256] f16
    const h16*   __restrict__ Ef,    // embedding^T frags [8][256][32][16]
    const float* __restrict__ b_out, // [4096]
    float*       __restrict__ out)   // [16384][4096] f32
{
  const int w    = threadIdx.x >> 5;
  const int lane = threadIdx.x & 31;
  const int nl   = lane & 15;
  const int hi   = lane >> 4;
  const int mt   = blockIdx.x;            // 0..1023
  const int nt   = blockIdx.y*8 + w;      // 0..255

  float bo = b_out[nt*16 + nl];
  v8f acc = {bo,bo,bo,bo,bo,bo,bo,bo};

  const h16* Abase = S + ((size_t)mt*16 + nl)*256 + (hi << 3);
  #pragma unroll
  for (int kt = 0; kt < 8; ++kt){
    const h16* p = Abase + kt*32;
    v8h lo = *(const v8h*)p;
    v8h hh = *(const v8h*)(p + 16);
    v16h a = __builtin_shufflevector(lo, hh, 0,1,2,3,4,5,6,7,8,9,10,11,12,13,14,15);
    v16h bf = *(const v16h*)(Ef + (((size_t)kt*256 + nt)*32 + lane)*16);
    acc = wmma16(a, bf, acc);
  }
  #pragma unroll
  for (int r = 0; r < 8; ++r)
    out[((size_t)mt*16 + r + 8*hi)*VOC + nt*16 + nl] = acc[r];
}

// ---------------------------------------------------------------------------
extern "C" void kernel_launch(void* const* d_in, const int* in_sizes, int n_in,
                              void* d_out, int out_size, void* d_ws, size_t ws_size,
                              hipStream_t stream) {
  const float* key      = (const float*)d_in[0];
  const float* values   = (const float*)d_in[1];
  const int*   lens     = (const int*)  d_in[2];  (void)lens;  // unused (no mask in ref)
  const int*   text     = (const int*)  d_in[3];
  const float* embedding= (const float*)d_in[4];
  const float* W_ih1    = (const float*)d_in[5];
  const float* W_hh1    = (const float*)d_in[6];
  const float* b_ih1    = (const float*)d_in[7];
  const float* b_hh1    = (const float*)d_in[8];
  const float* W_ih2    = (const float*)d_in[9];
  const float* W_hh2    = (const float*)d_in[10];
  const float* b_ih2    = (const float*)d_in[11];
  const float* b_hh2    = (const float*)d_in[12];
  const float* b_out    = (const float*)d_in[13];
  float* out = (float*)d_out;

  char* ws = (char*)d_ws;
  h16*   wih1f = (h16*)  (ws + OFF_WIH1);
  h16*   whh1f = (h16*)  (ws + OFF_WHH1);
  h16*   wih2f = (h16*)  (ws + OFF_WIH2);
  h16*   whh2f = (h16*)  (ws + OFF_WHH2);
  h16*   epkf  = (h16*)  (ws + OFF_EPK);
  float* b1    = (float*)(ws + OFF_B1);
  float* b2    = (float*)(ws + OFF_B2);
  h16*   Sbuf  = (h16*)  (ws + OFF_S);

  // phase 0: pack weights into WMMA B-fragment layout (f16)
  pack_b_frag<<<(12*64*512 + 255)/256, 256, 0, stream>>>(W_ih1, wih1f, 384, 1024);
  pack_b_frag<<<( 8*64*512 + 255)/256, 256, 0, stream>>>(W_hh1, whh1f, 256, 1024);
  pack_b_frag<<<( 8*32*512 + 255)/256, 256, 0, stream>>>(W_ih2, wih2f, 256,  512);
  pack_b_frag<<<( 4*32*512 + 255)/256, 256, 0, stream>>>(W_hh2, whh2f, 128,  512);
  pack_b_frag<<<(8*256*512 + 255)/256, 256, 0, stream>>>(embedding, epkf, 256, 4096);
  bias_combine<<<4, 256, 0, stream>>>(b_ih1, b_hh1, b_ih2, b_hh2, b1, b2);

  // phase 1: sequential recurrence, one workgroup per batch element
  decoder_recurrent<<<NB, 256, 0, stream>>>(key, values, text, embedding,
                                            wih1f, whh1f, wih2f, whh2f,
                                            b1, b2, Sbuf);

  // phase 2: massively parallel logits GEMM
  decoder_logits<<<dim3(1024, 32), 256, 0, stream>>>(Sbuf, epkf, b_out, out);
}
